// ResNetUNetDAMLastLayerv2_59313498358084
// MI455X (gfx1250) — matrix operand order repaired and checked by reading, not compiled, not run on hardware
//
#include <hip/hip_runtime.h>
#include <hip/hip_bf16.h>

// ---------------------------------------------------------------------------
// Dims: B=64, CIN=F=COUT=64, NC=27, NG=32, L=1024. EPS=1e-5.
// ---------------------------------------------------------------------------

typedef __attribute__((ext_vector_type(16))) _Float16 v16h;
typedef __attribute__((ext_vector_type(8)))  float    v8f;

#define LDH 72   // padded leading dim for f16 LDS tiles (144 B rows, 16B-aligned)

// A-fragment (16x32 f16, M x K) from row-major LDS tile, ld in halves.
__device__ __forceinline__ v16h frag_a(const _Float16* p, int ld) {
  int lane = threadIdx.x & 31;
  int m  = lane & 15;
  int kb = (lane >> 4) << 3;               // 0 or 8
  v16h a;
#pragma unroll
  for (int h = 0; h < 16; ++h) {
    int j = h >> 1;
    int k = kb + ((j >> 2) << 4) + ((j & 3) << 1) + (h & 1);
    a[h] = p[m * ld + k];
  }
  return a;
}

// B-fragment (32x16 f16, K x N): lane%16 = N column, halves = K (same order as frag_a).
__device__ __forceinline__ v16h frag_b(const _Float16* p, int ld) {
  int lane = threadIdx.x & 31;
  int n  = lane & 15;
  int kb = (lane >> 4) << 3;
  v16h b;
#pragma unroll
  for (int h = 0; h < 16; ++h) b[h] = p[(kb + ((h >> 3) << 4) + (h & 7)) * ld + n];
  return b;
}

__device__ __forceinline__ v8f wmma32(v16h a, v16h b, v8f c) {
  return __builtin_amdgcn_wmma_f32_16x16x32_f16(
      false, a, false, b, (short)0, c, false, false);
}

// ---------------------------------------------------------------------------
// K1: SPp = maxpool2x2(softmax_c(SP)); emits f32 and f16 copies.
// ---------------------------------------------------------------------------
__global__ void k_softmax_pool(const float* __restrict__ SP,
                               float* __restrict__ SPp,
                               _Float16* __restrict__ SPp_h) {
  int idx = blockIdx.x * blockDim.x + threadIdx.x;   // B*1024
  if (idx >= 64 * 1024) return;
  int b = idx >> 10;
  int p = idx & 1023;
  int oy = p >> 5, ox = p & 31;
  const float* base = SP + (size_t)b * 27 * 4096;
  float res[27];
#pragma unroll
  for (int c = 0; c < 27; ++c) res[c] = 0.0f;
#pragma unroll
  for (int q = 0; q < 4; ++q) {
    int y = oy * 2 + (q >> 1), x = ox * 2 + (q & 1);
    int off = y * 64 + x;
    float v[27];
    float m = -1e30f;
#pragma unroll
    for (int c = 0; c < 27; ++c) { v[c] = base[c * 4096 + off]; m = fmaxf(m, v[c]); }
    float s = 0.0f;
#pragma unroll
    for (int c = 0; c < 27; ++c) { v[c] = __expf(v[c] - m); s += v[c]; }
    float inv = 1.0f / s;
#pragma unroll
    for (int c = 0; c < 27; ++c) res[c] = fmaxf(res[c], v[c] * inv);
  }
  float* out = SPp + (size_t)b * 27 * 1024 + p;
  _Float16* outh = SPp_h + (size_t)b * 27 * 1024 + p;
#pragma unroll
  for (int c = 0; c < 27; ++c) { out[c * 1024] = res[c]; outh[c * 1024] = (_Float16)res[c]; }
}

// ---------------------------------------------------------------------------
// K2: f32 -> f16 converts for the shared GEMM operands (adj 1M + w_nc 27K).
// ---------------------------------------------------------------------------
__global__ void k_cvt(const float* __restrict__ adj, _Float16* __restrict__ adj_h,
                      const float* __restrict__ w_nc, _Float16* __restrict__ w_nc_h) {
  int i = blockIdx.x * blockDim.x + threadIdx.x;
  if (i < 1024 * 1024) adj_h[i] = (_Float16)adj[i];
  else if (i < 1024 * 1024 + 27 * 1024) {
    int j = i - 1024 * 1024;
    w_nc_h[j] = (_Float16)w_nc[j];
  }
}

// ---------------------------------------------------------------------------
// K3: t[b][f][l] = relu(sc1[f] * sum_c W[f][c] x[b][c][l] + b1[f])  (WMMA)
//     yt_h[b][l][f] stored via LDS bounce -> b128 stores.
// ---------------------------------------------------------------------------
__global__ void k_trans(const float* __restrict__ x,
                        const float* __restrict__ tw,
                        const float* __restrict__ g1,
                        const float* __restrict__ b1,
                        float* __restrict__ t,
                        _Float16* __restrict__ yt_h) {
  int b = blockIdx.y;
  int l0 = blockIdx.x * 64;
  __shared__ _Float16 sA[64 * LDH];
  __shared__ _Float16 sB[64 * LDH];
  int tid = threadIdx.x;
  for (int idx = tid; idx < 4096; idx += 256) {
    int f = idx >> 6, c = idx & 63;
    sA[f * LDH + c] = (_Float16)tw[f * 64 + c];
  }
  const float* xb = x + ((size_t)b * 64) * 1024 + l0;
  for (int idx = tid; idx < 4096; idx += 256) {
    int c = idx >> 6, l = idx & 63;
    sB[c * LDH + l] = (_Float16)xb[c * 1024 + l];
  }
  __syncthreads();
  int w = tid >> 5, lane = tid & 31;
  int mt = w & 3, nt0 = (w >> 2) << 1;
  v8f acc0 = {}, acc1 = {};
#pragma unroll
  for (int ks = 0; ks < 2; ++ks) {
    v16h a   = frag_a(&sA[(mt * 16) * LDH + ks * 32], LDH);
    v16h bb0 = frag_b(&sB[(ks * 32) * LDH + nt0 * 16], LDH);
    v16h bb1 = frag_b(&sB[(ks * 32) * LDH + (nt0 + 1) * 16], LDH);
    acc0 = wmma32(a, bb0, acc0);
    acc1 = wmma32(a, bb1, acc1);
  }
  __syncthreads();   // done reading sA/sB; reuse sA as yt bounce buffer
  int hi = (lane >> 4) << 3;
  int n = lane & 15;
  const float rs = rsqrtf(1.0f + 1e-5f);
  float* tb = t + ((size_t)b * 64) * 1024;
#pragma unroll
  for (int r = 0; r < 8; ++r) {
    int f = mt * 16 + r + hi;
    float sc = g1[f] * rs;
    float bb = b1[f];
    int ll = nt0 * 16 + n;
    float v = fmaxf(sc * acc0[r] + bb, 0.0f);
    tb[(size_t)f * 1024 + l0 + ll] = v;
    sA[ll * LDH + f] = (_Float16)v;
    ll = (nt0 + 1) * 16 + n;
    v = fmaxf(sc * acc1[r] + bb, 0.0f);
    tb[(size_t)f * 1024 + l0 + ll] = v;
    sA[ll * LDH + f] = (_Float16)v;
  }
  __syncthreads();
  _Float16* yb = yt_h + ((size_t)b * 1024 + l0) * 64;
  for (int seg = tid; seg < 512; seg += 256) {
    int r = seg >> 3, c8 = (seg & 7) * 8;
    *(uint4*)(yb + (size_t)r * 64 + c8) = *(const uint4*)(&sA[r * LDH + c8]);
  }
}

// ---------------------------------------------------------------------------
// K4: one 64x64x1024 WMMA GEMM per batch computes BOTH small projections:
//   A rows 0..26  = w_nc  ->  ynT[c][f] = sum_l w_nc[c][l] yt[l][f]
//   A rows 32..58 = SPp   ->  P  [c][f] = sum_l SPp[c][l]  yt[l][f]
// then sigma = w_kc . ynT (+biases), Q = sigma . P   (so A@yt = SPp^T Q).
// ---------------------------------------------------------------------------
__global__ void k_small(const _Float16* __restrict__ w_nc_h,
                        const _Float16* __restrict__ spp_h,
                        const _Float16* __restrict__ yt_h,
                        const float* __restrict__ b_nc,
                        const float* __restrict__ w_kc,
                        const float* __restrict__ b_kc,
                        float* __restrict__ Q) {
  int b = blockIdx.x;
  __shared__ _Float16 sA[64 * LDH];
  __shared__ _Float16 sB[64 * LDH];
  __shared__ float sYN[27 * 64];
  __shared__ float sP[27 * 64];
  __shared__ float sSig[27 * 27];
  int tid = threadIdx.x, w = tid >> 5, lane = tid & 31;
  int mt = w & 3, nt0 = (w >> 2) << 1;
  const _Float16* sph = spp_h + (size_t)b * 27 * 1024;
  const _Float16* yb  = yt_h + ((size_t)b * 1024) * 64;
  v8f acc0 = {}, acc1 = {};
  for (int k0 = 0; k0 < 1024; k0 += 64) {
    __syncthreads();
    for (int seg = tid; seg < 512; seg += 256) {
      int r = seg >> 3, c8 = (seg & 7) * 8;
      uint4 v = {0u, 0u, 0u, 0u};
      if (r < 27)                 v = *(const uint4*)(w_nc_h + (size_t)r * 1024 + k0 + c8);
      else if (r >= 32 && r < 59) v = *(const uint4*)(sph + (size_t)(r - 32) * 1024 + k0 + c8);
      *(uint4*)(&sA[r * LDH + c8]) = v;
      *(uint4*)(&sB[r * LDH + c8]) = *(const uint4*)(yb + (size_t)(k0 + r) * 64 + c8);
    }
    __syncthreads();
#pragma unroll
    for (int ks = 0; ks < 2; ++ks) {
      v16h a   = frag_a(&sA[(mt * 16) * LDH + ks * 32], LDH);
      v16h bb0 = frag_b(&sB[(ks * 32) * LDH + nt0 * 16], LDH);
      v16h bb1 = frag_b(&sB[(ks * 32) * LDH + (nt0 + 1) * 16], LDH);
      acc0 = wmma32(a, bb0, acc0);
      acc1 = wmma32(a, bb1, acc1);
    }
  }
  int hi = (lane >> 4) << 3, n = lane & 15;
#pragma unroll
  for (int r = 0; r < 8; ++r) {
    int m = mt * 16 + r + hi;
    int f0 = nt0 * 16 + n, f1 = f0 + 16;
    if (m < 27) {
      float bb = b_nc[m];
      sYN[m * 64 + f0] = acc0[r] + bb;
      sYN[m * 64 + f1] = acc1[r] + bb;
    } else if (m >= 32 && m < 59) {
      sP[(m - 32) * 64 + f0] = acc0[r];
      sP[(m - 32) * 64 + f1] = acc1[r];
    }
  }
  __syncthreads();
  for (int idx = tid; idx < 27 * 27; idx += 256) {
    int c = idx / 27, k = idx % 27;
    float s = 0.0f;
#pragma unroll
    for (int f = 0; f < 64; ++f) s += w_kc[k * 64 + f] * sYN[c * 64 + f];
    sSig[c * 27 + k] = s + b_kc[k];
  }
  __syncthreads();
  float* Qb = Q + (size_t)b * 27 * 64;
  for (int idx = tid; idx < 27 * 64; idx += 256) {
    int c = idx >> 6, f = idx & 63;
    float s = 0.0f;
#pragma unroll
    for (int k = 0; k < 27; ++k) s += sSig[c * 27 + k] * sP[k * 64 + f];
    Qb[idx] = s;
  }
}

// ---------------------------------------------------------------------------
// K5: sp_pre_h[b][l][f] = sum_m adj[l][m] yt[b][m][f]  (WMMA, K=1024)
//     b128 global loads, double-buffered LDS, prefetch 2 K-steps ahead,
//     f16 result via LDS bounce -> b128 stores.
// ---------------------------------------------------------------------------
__global__ void k_big(const _Float16* __restrict__ adj_h,
                      const _Float16* __restrict__ yt_h,
                      _Float16* __restrict__ sp_pre_h) {
  __shared__ _Float16 sA[2][64 * LDH];
  __shared__ _Float16 sB[2][64 * LDH];
  int b = blockIdx.y;
  int l0 = blockIdx.x * 64;
  int tid = threadIdx.x;
  int w = tid >> 5, lane = tid & 31;
  int mt = w & 3, nt0 = (w >> 2) << 1;
  const _Float16* yb = yt_h + ((size_t)b * 1024) * 64;

  int s0 = tid * 2, s1 = tid * 2 + 1;
  int r0 = s0 >> 3, c0 = (s0 & 7) * 8;
  int r1 = s1 >> 3, c1 = (s1 & 7) * 8;

  uint4 a0 = *(const uint4*)(adj_h + (size_t)(l0 + r0) * 1024 + c0);
  uint4 a1 = *(const uint4*)(adj_h + (size_t)(l0 + r1) * 1024 + c1);
  uint4 b0 = *(const uint4*)(yb + (size_t)r0 * 64 + c0);
  uint4 b1 = *(const uint4*)(yb + (size_t)r1 * 64 + c1);
  *(uint4*)(&sA[0][r0 * LDH + c0]) = a0;
  *(uint4*)(&sA[0][r1 * LDH + c1]) = a1;
  *(uint4*)(&sB[0][r0 * LDH + c0]) = b0;
  *(uint4*)(&sB[0][r1 * LDH + c1]) = b1;
  __syncthreads();

  v8f acc0 = {}, acc1 = {};
  for (int it = 0; it < 16; ++it) {
    int cur = it & 1, nxt = cur ^ 1;
    if (it < 15) {
      int k0 = (it + 1) * 64;
      a0 = *(const uint4*)(adj_h + (size_t)(l0 + r0) * 1024 + k0 + c0);
      a1 = *(const uint4*)(adj_h + (size_t)(l0 + r1) * 1024 + k0 + c1);
      b0 = *(const uint4*)(yb + (size_t)(k0 + r0) * 64 + c0);
      b1 = *(const uint4*)(yb + (size_t)(k0 + r1) * 64 + c1);
      if (it < 14) {  // pull K-step it+2 toward the WGP (global_prefetch_b8)
        __builtin_prefetch(adj_h + (size_t)(l0 + r0) * 1024 + k0 + 64 + c0, 0, 0);
        __builtin_prefetch(yb + (size_t)(k0 + 64 + r0) * 64 + c0, 0, 0);
      }
    }
#pragma unroll
    for (int ks = 0; ks < 2; ++ks) {
      v16h a   = frag_a(&sA[cur][(mt * 16) * LDH + ks * 32], LDH);
      v16h bb0 = frag_b(&sB[cur][(ks * 32) * LDH + nt0 * 16], LDH);
      v16h bb1 = frag_b(&sB[cur][(ks * 32) * LDH + (nt0 + 1) * 16], LDH);
      acc0 = wmma32(a, bb0, acc0);
      acc1 = wmma32(a, bb1, acc1);
    }
    if (it < 15) {
      *(uint4*)(&sA[nxt][r0 * LDH + c0]) = a0;
      *(uint4*)(&sA[nxt][r1 * LDH + c1]) = a1;
      *(uint4*)(&sB[nxt][r0 * LDH + c0]) = b0;
      *(uint4*)(&sB[nxt][r1 * LDH + c1]) = b1;
    }
    __syncthreads();
  }
  int hi = (lane >> 4) << 3, n = lane & 15;
#pragma unroll
  for (int r = 0; r < 8; ++r) {
    int lm = mt * 16 + r + hi;
    sA[0][lm * LDH + nt0 * 16 + n] = (_Float16)acc0[r];
    sA[0][lm * LDH + (nt0 + 1) * 16 + n] = (_Float16)acc1[r];
  }
  __syncthreads();
  _Float16* spb = sp_pre_h + ((size_t)b * 1024 + l0) * 64;
  *(uint4*)(spb + (size_t)r0 * 64 + c0) = *(const uint4*)(&sA[0][r0 * LDH + c0]);
  *(uint4*)(spb + (size_t)r1 * 64 + c1) = *(const uint4*)(&sA[0][r1 * LDH + c1]);
}

// ---------------------------------------------------------------------------
// K6: fused epilogue per (batch, 64 l-rows):
//   se_pre = SPp^T Q (K=27, VALU) -> GEMM w_g^T -> relu(+b_g)
//   + GEMM(sp_pre_h, w_sp^T) -> relu(+b_sp); + 3*t (all residuals folded)
//   -> GEMM back_w^T -> BN+relu -> coalesced store via padded LDS.
// ---------------------------------------------------------------------------
__global__ void k_epi(const _Float16* __restrict__ sp_pre_h,
                      const float* __restrict__ SPp,
                      const float* __restrict__ Q,
                      const float* __restrict__ t,
                      const float* __restrict__ w_g,
                      const float* __restrict__ b_g,
                      const float* __restrict__ w_sp,
                      const float* __restrict__ b_sp,
                      const float* __restrict__ back_w,
                      const float* __restrict__ g2,
                      const float* __restrict__ b2,
                      float* __restrict__ out) {
  int b = blockIdx.y;
  int l0 = blockIdx.x * 64;
  __shared__ _Float16 sA[64 * LDH];
  __shared__ _Float16 sB[64 * LDH];
  __shared__ float sAcc[64 * 65];
  __shared__ float sQ[27 * 64];
  __shared__ float sSP[27 * 64];
  int tid = threadIdx.x;
  int w = tid >> 5, lane = tid & 31;
  int mt = w & 3, nt0 = (w >> 2) << 1;
  int hi = (lane >> 4) << 3, n = lane & 15;

  for (int idx = tid; idx < 27 * 64; idx += 256) {
    sQ[idx] = Q[(size_t)b * 27 * 64 + idx];
    int c = idx >> 6, l = idx & 63;
    sSP[c * 64 + l] = SPp[((size_t)b * 27 + c) * 1024 + l0 + l];
  }
  for (int idx = tid; idx < 4096; idx += 256) {
    int f = idx >> 6, g = idx & 63;
    sB[g * LDH + f] = (_Float16)w_g[f * 64 + g];     // sB[g][f] = w_g[f][g]
  }
  __syncthreads();
  for (int idx = tid; idx < 4096; idx += 256) {       // se_pre -> sA (f16)
    int l = idx & 63, f = idx >> 6;
    float s = 0.0f;
#pragma unroll
    for (int c = 0; c < 27; ++c) s += sSP[c * 64 + l] * sQ[c * 64 + f];
    sA[l * LDH + f] = (_Float16)s;
  }
  __syncthreads();

  {  // GEMM1: se_pre @ w_g^T -> relu(+b_g) -> sAcc
    v8f a0 = {}, a1 = {};
#pragma unroll
    for (int ks = 0; ks < 2; ++ks) {
      v16h a   = frag_a(&sA[(mt * 16) * LDH + ks * 32], LDH);
      v16h bb0 = frag_b(&sB[(ks * 32) * LDH + nt0 * 16], LDH);
      v16h bb1 = frag_b(&sB[(ks * 32) * LDH + (nt0 + 1) * 16], LDH);
      a0 = wmma32(a, bb0, a0);
      a1 = wmma32(a, bb1, a1);
    }
#pragma unroll
    for (int r = 0; r < 8; ++r) {
      int ll = mt * 16 + r + hi;
      int f0 = nt0 * 16 + n, f1 = f0 + 16;
      sAcc[ll * 65 + f0] = fmaxf(a0[r] + b_g[f0], 0.0f);
      sAcc[ll * 65 + f1] = fmaxf(a1[r] + b_g[f1], 0.0f);
    }
  }
  __syncthreads();

  const _Float16* spb = sp_pre_h + ((size_t)b * 1024 + l0) * 64;
  for (int seg = tid; seg < 512; seg += 256) {        // sp_pre -> sA (b128)
    int r = seg >> 3, c8 = (seg & 7) * 8;
    *(uint4*)(&sA[r * LDH + c8]) = *(const uint4*)(spb + (size_t)r * 64 + c8);
  }
  for (int idx = tid; idx < 4096; idx += 256) {
    int f = idx >> 6, g = idx & 63;
    sB[g * LDH + f] = (_Float16)w_sp[f * 64 + g];
  }
  __syncthreads();

  {  // GEMM2: sp_pre @ w_sp^T -> relu(+b_sp), accumulate into sAcc
    v8f a0 = {}, a1 = {};
#pragma unroll
    for (int ks = 0; ks < 2; ++ks) {
      v16h a   = frag_a(&sA[(mt * 16) * LDH + ks * 32], LDH);
      v16h bb0 = frag_b(&sB[(ks * 32) * LDH + nt0 * 16], LDH);
      v16h bb1 = frag_b(&sB[(ks * 32) * LDH + (nt0 + 1) * 16], LDH);
      a0 = wmma32(a, bb0, a0);
      a1 = wmma32(a, bb1, a1);
    }
#pragma unroll
    for (int r = 0; r < 8; ++r) {
      int ll = mt * 16 + r + hi;
      int f0 = nt0 * 16 + n, f1 = f0 + 16;
      sAcc[ll * 65 + f0] += fmaxf(a0[r] + b_sp[f0], 0.0f);
      sAcc[ll * 65 + f1] += fmaxf(a1[r] + b_sp[f1], 0.0f);
    }
  }
  __syncthreads();

  const float* tb = t + ((size_t)b * 64) * 1024 + l0;
  for (int idx = tid; idx < 4096; idx += 256) {       // + 3*t -> sA (f16)
    int l = idx & 63, f = idx >> 6;
    float v = sAcc[l * 65 + f] + 3.0f * tb[(size_t)f * 1024 + l];
    sA[l * LDH + f] = (_Float16)v;
  }
  for (int idx = tid; idx < 4096; idx += 256) {
    int o = idx >> 6, f = idx & 63;
    sB[f * LDH + o] = (_Float16)back_w[o * 64 + f];
  }
  __syncthreads();

  {  // GEMM3: pre @ back_w^T -> BN + relu -> sAcc (out^T tile)
    v8f a0 = {}, a1 = {};
#pragma unroll
    for (int ks = 0; ks < 2; ++ks) {
      v16h a   = frag_a(&sA[(mt * 16) * LDH + ks * 32], LDH);
      v16h bb0 = frag_b(&sB[(ks * 32) * LDH + nt0 * 16], LDH);
      v16h bb1 = frag_b(&sB[(ks * 32) * LDH + (nt0 + 1) * 16], LDH);
      a0 = wmma32(a, bb0, a0);
      a1 = wmma32(a, bb1, a1);
    }
    const float rs = rsqrtf(1.0f + 1e-5f);
#pragma unroll
    for (int r = 0; r < 8; ++r) {
      int ll = mt * 16 + r + hi;
      int o0 = nt0 * 16 + n, o1 = o0 + 16;
      sAcc[ll * 65 + o0] = fmaxf(g2[o0] * rs * a0[r] + b2[o0], 0.0f);
      sAcc[ll * 65 + o1] = fmaxf(g2[o1] * rs * a1[r] + b2[o1], 0.0f);
    }
  }
  __syncthreads();

  float* ob = out + ((size_t)b * 64) * 1024 + l0;
  for (int idx = tid; idx < 4096; idx += 256) {
    int l = idx & 63, o = idx >> 6;
    ob[(size_t)o * 1024 + l] = sAcc[l * 65 + o];
  }
}

// ---------------------------------------------------------------------------
extern "C" void kernel_launch(void* const* d_in, const int* in_sizes, int n_in,
                              void* d_out, int out_size, void* d_ws, size_t ws_size,
                              hipStream_t stream) {
  const float* x       = (const float*)d_in[0];
  const float* SP      = (const float*)d_in[1];
  const float* trans_w = (const float*)d_in[2];
  const float* bn1_g   = (const float*)d_in[3];
  const float* bn1_b   = (const float*)d_in[4];
  const float* w_nc    = (const float*)d_in[5];
  const float* b_nc    = (const float*)d_in[6];
  const float* w_kc    = (const float*)d_in[7];
  const float* b_kc    = (const float*)d_in[8];
  const float* w_g     = (const float*)d_in[9];
  const float* b_g     = (const float*)d_in[10];
  const float* adj_sp  = (const float*)d_in[11];
  const float* w_sp    = (const float*)d_in[12];
  const float* b_sp    = (const float*)d_in[13];
  const float* back_w  = (const float*)d_in[14];
  const float* bn2_g   = (const float*)d_in[15];
  const float* bn2_b   = (const float*)d_in[16];

  char* ws = (char*)d_ws;
  size_t o = 0;
  float*    SPp      = (float*)(ws + o);    o += 64ull * 27 * 1024 * 4;   // 7.1 MB
  _Float16* SPp_h    = (_Float16*)(ws + o); o += 64ull * 27 * 1024 * 2;   // 3.5 MB
  float*    t        = (float*)(ws + o);    o += 64ull * 64 * 1024 * 4;   // 16.8 MB
  _Float16* yt_h     = (_Float16*)(ws + o); o += 64ull * 1024 * 64 * 2;   // 8.4 MB
  _Float16* adj_h    = (_Float16*)(ws + o); o += 1024ull * 1024 * 2;      // 2.1 MB
  _Float16* w_nc_h   = (_Float16*)(ws + o); o += 27ull * 1024 * 2;        // 55 KB
  _Float16* sp_pre_h = (_Float16*)(ws + o); o += 64ull * 1024 * 64 * 2;   // 8.4 MB
  float*    Qb       = (float*)(ws + o);    o += 64ull * 27 * 64 * 4;     // 0.4 MB

  k_softmax_pool<<<256, 256, 0, stream>>>(SP, SPp, SPp_h);
  k_cvt<<<4204, 256, 0, stream>>>(adj_sp, adj_h, w_nc, w_nc_h);
  k_trans<<<dim3(16, 64), 256, 0, stream>>>(x, trans_w, bn1_g, bn1_b, t, yt_h);
  k_small<<<64, 256, 0, stream>>>(w_nc_h, SPp_h, yt_h, b_nc, w_kc, b_kc, Qb);
  k_big<<<dim3(16, 64), 256, 0, stream>>>(adj_h, yt_h, sp_pre_h);
  k_epi<<<dim3(16, 64), 256, 0, stream>>>(sp_pre_h, SPp, Qb, t, w_g, b_g,
                                          w_sp, b_sp, back_w, bn2_g, bn2_b,
                                          (float*)d_out);
}
